// Attention_1537598292670
// MI455X (gfx1250) — matrix operand order repaired and checked by reading, not compiled, not run on hardware
//
#include <hip/hip_runtime.h>

// ---------------------------------------------------------------------------
// Flash-attention forward for: B=16, TQ=TK=2048, H=512, fp32 in/out.
// bf16 WMMA (v_wmma_f32_16x16x32_bf16) with f32 accumulation, online softmax.
// ---------------------------------------------------------------------------

typedef __attribute__((ext_vector_type(16))) __bf16 v16bf;
typedef __attribute__((ext_vector_type(2)))  __bf16 v2bf;
typedef __attribute__((ext_vector_type(8)))  float  v8f;

#define NB    16
#define TQ    2048
#define TKV   2048
#define HD    512

#define QBLK  64      // Q rows per workgroup
#define KBLK  32      // K/V rows per iteration
#define NWAVE 8

// LDS strides in bf16 elements (padded to break bank conflicts)
#define QS   (HD + 8)     // 520  (1040 B, 16B aligned)
#define KS   (HD + 8)     // 520
#define VTS  (KBLK + 8)   // 40   (80 B, 16B aligned)
#define PS   (KBLK + 8)   // 40

#define LDS_Q_ELEMS   (QBLK * QS)          // 33280
#define LDS_K_ELEMS   (KBLK * KS)          // 16640
#define LDS_VT_ELEMS  (HD * VTS)           // 20480
#define LDS_P_ELEMS   (NWAVE * 16 * PS)    // 5120
#define LDS_TOTAL_BYTES ((LDS_Q_ELEMS + LDS_K_ELEMS + LDS_VT_ELEMS + LDS_P_ELEMS) * 2)

// two f32 -> packed bf16x2 (hardware cvt; RNE)
static __device__ __forceinline__ unsigned int pk_bf16(float x, float y) {
    v2bf v;
    v[0] = (__bf16)x;
    v[1] = (__bf16)y;
    return __builtin_bit_cast(unsigned int, v);
}

// single f32 -> bf16 bits (hardware cvt; RNE)
static __device__ __forceinline__ unsigned short bf1(float x) {
    return __builtin_bit_cast(unsigned short, (__bf16)x);
}

static __device__ __forceinline__ v16bf load_frag(const unsigned short* p0,
                                                  const unsigned short* p1) {
    v16bf a;
    ((uint4*)&a)[0] = *(const uint4*)p0;
    ((uint4*)&a)[1] = *(const uint4*)p1;
    return a;
}

__global__ __launch_bounds__(256, 1)
void fa_fwd_kernel(const float* __restrict__ ys,
                   const float* __restrict__ hs,
                   float* __restrict__ out)
{
    extern __shared__ char smem_raw[];
    unsigned short* Qls  = (unsigned short*)smem_raw;
    unsigned short* Kls  = Qls + LDS_Q_ELEMS;
    unsigned short* VTls = Kls + LDS_K_ELEMS;
    unsigned short* Pls  = VTls + LDS_VT_ELEMS;

    const int tid   = threadIdx.x;
    const int lane  = tid & 31;
    const int wave  = tid >> 5;
    const int mt    = wave >> 1;          // 0..3 : 16-row Q sub-tile
    const int hh    = wave & 1;           // 0..1 : 256-wide H half for P·V
    const int ln16  = lane & 15;
    const int hi16  = (lane >> 4) & 1;    // lane half

    const int b  = blockIdx.y;
    const int q0 = blockIdx.x * QBLK;

    // ---- stage Q block: f32 -> bf16 into LDS (rows contiguous in global) ----
    {
        const float* qsrc = ys + ((size_t)b * TQ + q0) * HD;
        #pragma unroll
        for (int i = 0; i < (QBLK * HD) / (256 * 4); ++i) {
            int idx = (i * 256 + tid) * 4;
            int r = idx >> 9;          // / HD
            int c = idx & (HD - 1);
            float4 f = *(const float4*)(qsrc + idx);
            uint2 pk;
            pk.x = pk_bf16(f.x, f.y);
            pk.y = pk_bf16(f.z, f.w);
            *(uint2*)(Qls + r * QS + c) = pk;
        }
    }

    // ---- per-wave accumulator state ----
    v8f o[16];
    #pragma unroll
    for (int t = 0; t < 16; ++t) o[t] = (v8f){0.f,0.f,0.f,0.f,0.f,0.f,0.f,0.f};
    float m[8], l[8];
    #pragma unroll
    for (int r = 0; r < 8; ++r) { m[r] = -3.4e38f; l[r] = 0.f; }

    const int arow = mt * 16 + ln16;                      // Q row this lane feeds
    unsigned short* Pw = Pls + wave * 16 * PS;            // per-wave P tile

    const float* hsb = hs + (size_t)b * TKV * HD;

    for (int kb = 0; kb < TKV / KBLK; ++kb) {
        __syncthreads();   // prior-iteration LDS reads done before overwrite

        // ---- stage K/V block: row-major bf16 + transposed bf16 copies ----
        const float* ksrc = hsb + (size_t)kb * KBLK * HD;
        #pragma unroll
        for (int i = 0; i < (KBLK * HD) / (256 * 4); ++i) {
            int idx = (i * 256 + tid) * 4;
            int r = idx >> 9;
            int c = idx & (HD - 1);
            float4 f = *(const float4*)(ksrc + idx);
            uint2 pk;
            pk.x = pk_bf16(f.x, f.y);
            pk.y = pk_bf16(f.z, f.w);
            *(uint2*)(Kls + r * KS + c) = pk;
            // transposed copy for P·V B-fragments (b16 stores ignore MSBs)
            VTls[(c + 0) * VTS + r] = (unsigned short)(pk.x);
            VTls[(c + 1) * VTS + r] = (unsigned short)(pk.x >> 16);
            VTls[(c + 2) * VTS + r] = (unsigned short)(pk.y);
            VTls[(c + 3) * VTS + r] = (unsigned short)(pk.y >> 16);
        }

        // prefetch next K/V block into cache hierarchy (global_prefetch_b8)
        if (kb + 1 < TKV / KBLK) {
            const char* nxt = (const char*)(ksrc + (size_t)KBLK * HD);
            __builtin_prefetch(nxt + tid * 256, 0, 0);
            __builtin_prefetch(nxt + tid * 256 + 128, 0, 0);
        }

        __syncthreads();

        // ---- S = Q · K^T over H, two 16-wide N tiles (K cols 0..31) ----
        v8f s0 = (v8f){0.f,0.f,0.f,0.f,0.f,0.f,0.f,0.f};
        v8f s1 = s0;
        #pragma unroll
        for (int hc = 0; hc < HD / 32; ++hc) {
            // A fragment of Q (16x32 bf16)
            const unsigned short* ap = Qls + arow * QS + hc * 32 + hi16 * 16;
            v16bf a = load_frag(ap, ap + 8);
            // B fragments of K^T (32x16 bf16), tiles n=0 and n=1
            const unsigned short* bp0 = Kls + ln16 * KS        + hc * 32 + hi16 * 16;
            const unsigned short* bp1 = Kls + (16 + ln16) * KS + hc * 32 + hi16 * 16;
            v16bf b0 = load_frag(bp0, bp0 + 8);
            v16bf b1 = load_frag(bp1, bp1 + 8);
            s0 = __builtin_amdgcn_wmma_f32_16x16x32_bf16(false, a, false, b0,
                                                         (short)0, s0, false, false);
            // same A operand as previous identical instruction -> reuse hint
            s1 = __builtin_amdgcn_wmma_f32_16x16x32_bf16(false, a, false, b1,
                                                         (short)0, s1, true, false);
        }

        // ---- online softmax (rows live per-VGPR; 16-lane half reductions) ----
        float corr[8];
        #pragma unroll
        for (int r = 0; r < 8; ++r) {
            float v = fmaxf(s0[r], s1[r]);
            v = fmaxf(v, __shfl_xor(v, 1, 32));
            v = fmaxf(v, __shfl_xor(v, 2, 32));
            v = fmaxf(v, __shfl_xor(v, 4, 32));
            v = fmaxf(v, __shfl_xor(v, 8, 32));
            float mnew = fmaxf(m[r], v);
            corr[r] = __expf(m[r] - mnew);
            m[r] = mnew;
            float p0 = __expf(s0[r] - mnew);
            float p1 = __expf(s1[r] - mnew);
            s0[r] = p0; s1[r] = p1;
            float rs = p0 + p1;
            rs += __shfl_xor(rs, 1, 32);
            rs += __shfl_xor(rs, 2, 32);
            rs += __shfl_xor(rs, 4, 32);
            rs += __shfl_xor(rs, 8, 32);
            l[r] = l[r] * corr[r] + rs;
        }

        // rescale O accumulators
        #pragma unroll
        for (int t = 0; t < 16; ++t)
            #pragma unroll
            for (int r = 0; r < 8; ++r)
                o[t][r] *= corr[r];

        // ---- P (bf16) -> per-wave LDS tile (C-layout -> row-major) ----
        #pragma unroll
        for (int r = 0; r < 8; ++r) {
            int row = r + hi16 * 8;
            Pw[row * PS + ln16]      = bf1(s0[r]);
            Pw[row * PS + 16 + ln16] = bf1(s1[r]);
        }
        // DS ops are in-order within a wave; re-read as A fragment (16x32)
        const unsigned short* pp = Pw + ln16 * PS + hi16 * 16;
        v16bf pa = load_frag(pp, pp + 8);

        // ---- O += P · V on this wave's 256-wide H half ----
        // t = 0 peeled: reuse flag must be a literal constant
        {
            int hcol = hh * 256 + ln16;
            const unsigned short* vp = VTls + hcol * VTS + hi16 * 16;
            v16bf vb = load_frag(vp, vp + 8);
            o[0] = __builtin_amdgcn_wmma_f32_16x16x32_bf16(false, pa, false, vb,
                                                           (short)0, o[0],
                                                           false, false);
        }
        #pragma unroll
        for (int t = 1; t < 16; ++t) {
            int hcol = hh * 256 + t * 16 + ln16;   // B column = lane's h col
            const unsigned short* vp = VTls + hcol * VTS + hi16 * 16;
            v16bf vb = load_frag(vp, vp + 8);
            // pa shared by the whole chain -> reuse hint after the first
            o[t] = __builtin_amdgcn_wmma_f32_16x16x32_bf16(false, pa, false, vb,
                                                           (short)0, o[t],
                                                           true, false);
        }
    }

    // ---- normalize and write out ----
    float* op = out + ((size_t)b * TQ + q0 + mt * 16) * HD;
    #pragma unroll
    for (int r = 0; r < 8; ++r) {
        int row = r + hi16 * 8;
        float inv = 1.0f / l[r];
        #pragma unroll
        for (int t = 0; t < 16; ++t) {
            int hcol = hh * 256 + t * 16 + ln16;
            op[(size_t)row * HD + hcol] = o[t][r] * inv;
        }
    }
}

extern "C" void kernel_launch(void* const* d_in, const int* in_sizes, int n_in,
                              void* d_out, int out_size, void* d_ws, size_t ws_size,
                              hipStream_t stream) {
    (void)in_sizes; (void)n_in; (void)out_size; (void)d_ws; (void)ws_size;
    const float* ys = (const float*)d_in[0];
    const float* hs = (const float*)d_in[1];
    float* out = (float*)d_out;

    // allow >64KB dynamic LDS (no-op if already permitted)
    (void)hipFuncSetAttribute((const void*)fa_fwd_kernel,
                              hipFuncAttributeMaxDynamicSharedMemorySize,
                              (int)LDS_TOTAL_BYTES);

    dim3 grid(TQ / QBLK, NB);
    dim3 block(256);
    fa_fwd_kernel<<<grid, block, LDS_TOTAL_BYTES, stream>>>(ys, hs, out);
}
